// MotionMambaBlock_78726750536309
// MI455X (gfx1250) — matrix operand order repaired and checked by reading, not compiled, not run on hardware
//
#include <hip/hip_runtime.h>
#include <hip/hip_bf16.h>

// ---------------------------------------------------------------------------
// MotionMamba block for MI455X (gfx1250, wave32, WMMA).
// GEMMs use v_wmma_f32_16x16x32_f16 with multi-accumulator tiles (A-fragment
// reuse across 3-4 N tiles). Selective scan keeps the 16-wide SSM state per
// (b,d) lane in VGPRs; dt/B/C rows are staged to LDS with
// GLOBAL_LOAD_ASYNC_TO_LDS_B128 (ASYNCcnt) in 32-step chunks.
// ---------------------------------------------------------------------------

typedef __attribute__((ext_vector_type(16))) _Float16 v16h;
typedef __attribute__((ext_vector_type(8)))  _Float16 v8h;
typedef __attribute__((ext_vector_type(8)))  float    v8f;
typedef __attribute__((ext_vector_type(4)))  int      v4i;

#define NB    16      // batch
#define NT    128     // d_temporal (features per step)
#define NL    1024    // scan length (= d_model)
#define DI    256     // d_inner
#define E2    512     // 2*d_inner
#define NST   16      // d_state
#define RNK   8       // dt_rank
#define RPAD  48      // x_dbl row padded (40 -> 48)

#if defined(__has_builtin)
#  if __has_builtin(__builtin_amdgcn_global_load_async_to_lds_b128)
#    define HAVE_ASYNC_LDS 1
#  else
#    define HAVE_ASYNC_LDS 0
#  endif
#else
#  define HAVE_ASYNC_LDS 0
#endif

// ------------------------- workspace layout (bytes) ------------------------
#define OFF_WIN16   ((size_t)0)                               // 512*128 f16
#define OFF_WXPF16  (OFF_WIN16  + (size_t)E2*NT*2)            // 48*256 f16
#define OFF_WXPB16  (OFF_WXPF16 + (size_t)RPAD*DI*2)
#define OFF_WOUT16  (OFF_WXPB16 + (size_t)RPAD*DI*2)          // 128*256 f16
#define OFF_HN16    (OFF_WOUT16 + (size_t)NT*DI*2)            // [b][l][128] f16
#define OFF_XZ16    (OFF_HN16   + (size_t)NB*NL*NT*2)         // [b][e][l] f16
#define OFF_XC16    (OFF_XZ16   + (size_t)NB*E2*NL*2)         // [b][l][256] f16
#define OFF_XC32    (OFF_XC16   + (size_t)NB*NL*DI*2)         // [b][l][256] f32
#define OFF_XDBL    (OFF_XC32   + (size_t)NB*NL*DI*4)         // [b][l][48]  f32
#define OFF_YACC    (OFF_XDBL   + (size_t)NB*NL*RPAD*4)       // [b][l][256] f32
#define OFF_YSUM16  (OFF_YACC   + (size_t)NB*NL*DI*4)         // [b][l][256] f16
#define OFF_G3      (OFF_YSUM16 + (size_t)NB*NL*DI*2)         // [b][l][128] f32

// ------------------------- WMMA fragment helpers ---------------------------
// A fragment (16x32, f16) from row-major [M][K] memory.
// lane m = lane&15 holds row m; chunks (lane>>4)*8 and 16+(lane>>4)*8.
__device__ __forceinline__ v16h load_a_frag(const _Float16* base, int lda,
                                            int m0, int k0, int lane) {
  const int m  = m0 + (lane & 15);
  const int ks = k0 + ((lane >> 4) << 3);
  const _Float16* p = base + (size_t)m * lda + ks;
  v8h lo = *(const v8h*)(p);
  v8h hi = *(const v8h*)(p + 16);
  v16h r;
#pragma unroll
  for (int i = 0; i < 8; ++i) { r[i] = lo[i]; r[i + 8] = hi[i]; }
  return r;
}

// B fragment (32x16, f16) from row-major [N][K] weights.
// lane n = lane&15 holds column n; chunks (lane>>4)*8 and 16+(lane>>4)*8.
__device__ __forceinline__ v16h load_b_frag(const _Float16* base, int ldb,
                                            int n0, int k0, int lane) {
  const int n  = n0 + (lane & 15);
  const int ks = k0 + ((lane >> 4) << 3);
  const _Float16* p = base + (size_t)n * ldb + ks;
  v8h lo = *(const v8h*)(p);
  v8h hi = *(const v8h*)(p + 16);
  v16h r;
#pragma unroll
  for (int i = 0; i < 8; ++i) { r[i] = lo[i]; r[i + 8] = hi[i]; }
  return r;
}

__device__ __forceinline__ void wmma_acc(v8f& c, v16h a, v16h b) {
  c = __builtin_amdgcn_wmma_f32_16x16x32_f16(false, a, false, b,
                                             (short)0, c, false, false);
}

// --------------------------- weight conversion -----------------------------
__global__ void mamba_cvt_weights(const float* win, const float* xpf,
                                  const float* xpb, const float* wout,
                                  _Float16* win16, _Float16* xpf16,
                                  _Float16* xpb16, _Float16* wout16) {
  const int NWIN = E2 * NT;        // 65536
  const int NXP  = RPAD * DI;      // 12288 (padded)
  const int NWO  = NT * DI;        // 32768
  const int total = NWIN + 2 * NXP + NWO;
  for (int i = blockIdx.x * blockDim.x + threadIdx.x; i < total;
       i += gridDim.x * blockDim.x) {
    if (i < NWIN) {
      win16[i] = (_Float16)win[i];
    } else if (i < NWIN + NXP) {
      int j = i - NWIN;  int r = j / DI;
      xpf16[j] = (r < RNK + 2 * NST) ? (_Float16)xpf[j] : (_Float16)0.0f;
    } else if (i < NWIN + 2 * NXP) {
      int j = i - NWIN - NXP;  int r = j / DI;
      xpb16[j] = (r < RNK + 2 * NST) ? (_Float16)xpb[j] : (_Float16)0.0f;
    } else {
      int j = i - NWIN - 2 * NXP;
      wout16[j] = (_Float16)wout[j];
    }
  }
}

// ----------------- input rmsnorm + transpose to f16 [b][l][t] --------------
__global__ void mamba_rmsnorm_in(const float* __restrict__ hid,
                                 const float* __restrict__ nw,
                                 _Float16* __restrict__ hn) {
  const int l = blockIdx.x & (NL - 1);
  const int b = blockIdx.x >> 10;
  const int t = threadIdx.x;                       // 128 threads
  float v = hid[((size_t)b * NT + t) * NL + l];
  float ss = v * v;
#pragma unroll
  for (int o = 16; o > 0; o >>= 1) ss += __shfl_xor(ss, o, 32);
  __shared__ float sred[4];
  if ((t & 31) == 0) sred[t >> 5] = ss;
  __syncthreads();
  float tot = sred[0] + sred[1] + sred[2] + sred[3];
  float inv = rsqrtf(tot * (1.0f / NT) + 1e-5f);
  hn[((size_t)b * NL + l) * NT + t] = (_Float16)(v * inv * nw[t]);
}

// -------- GEMM1: xz[b][e][l] = hn[b][l][:] . in_proj_w[e][:]  (K=128) ------
// One wave -> 1 M-tile x 4 N-tiles (A fragment reused 4x per k-step).
__global__ void mamba_gemm_inproj(const _Float16* __restrict__ hn,
                                  const _Float16* __restrict__ win,
                                  _Float16* __restrict__ xz) {
  const int wave = blockIdx.x * 8 + (threadIdx.x >> 5);   // 16*64*8 waves
  const int lane = threadIdx.x & 31;
  const int b  = wave / (64 * 8);
  const int rr = wave % (64 * 8);
  const int lt = rr / 8, eg = rr % 8;                     // eg: group of 4 e-tiles
  const _Float16* A = hn + (size_t)b * NL * NT;
  v8f c[4] = {};
#pragma unroll
  for (int k = 0; k < NT; k += 32) {
    v16h a = load_a_frag(A, NT, lt * 16, k, lane);
#pragma unroll
    for (int j = 0; j < 4; ++j) {
      v16h bb = load_b_frag(win, NT, (eg * 4 + j) * 16, k, lane);
      wmma_acc(c[j], a, bb);
    }
  }
  const int n = lane & 15, mb = (lane >> 4) * 8;
#pragma unroll
  for (int j = 0; j < 4; ++j) {
    _Float16* dst =
        xz + ((size_t)b * E2 + (eg * 4 + j) * 16 + n) * NL + lt * 16 + mb;
    v8h o;
#pragma unroll
    for (int i = 0; i < 8; ++i) o[i] = (_Float16)c[j][i];
    *(v8h*)dst = o;
  }
}

// ---------------- depthwise causal/anticausal conv + SiLU ------------------
__global__ void mamba_conv_silu(const _Float16* __restrict__ xz,
                                const float* __restrict__ cw,
                                const float* __restrict__ cb, int dir,
                                _Float16* __restrict__ xc16,
                                float* __restrict__ xc32) {
  const int d = blockIdx.x & (DI - 1);
  const int b = blockIdx.x >> 8;
  const _Float16* x = xz + ((size_t)b * E2 + d) * NL;      // x half of xz
  const float w0 = cw[d * 4], w1 = cw[d * 4 + 1];
  const float w2 = cw[d * 4 + 2], w3 = cw[d * 4 + 3];
  const float bias = cb[d];
  for (int l = threadIdx.x; l < NL; l += blockDim.x) {
    float acc = bias;
    if (dir == 0) {  // causal: sum_k w[k] * x[l-3+k]
      acc += w3 * (float)x[l];
      if (l >= 1) acc += w2 * (float)x[l - 1];
      if (l >= 2) acc += w1 * (float)x[l - 2];
      if (l >= 3) acc += w0 * (float)x[l - 3];
    } else {        // reversed stream: sum_k w[k] * x[l+3-k]
      acc += w3 * (float)x[l];
      if (l + 1 < NL) acc += w2 * (float)x[l + 1];
      if (l + 2 < NL) acc += w1 * (float)x[l + 2];
      if (l + 3 < NL) acc += w0 * (float)x[l + 3];
    }
    const float s = acc / (1.0f + __expf(-acc));           // SiLU
    const size_t oi = ((size_t)b * NL + l) * DI + d;
    xc32[oi] = s;
    xc16[oi] = (_Float16)s;
  }
}

// ------ GEMM2: x_dbl[b][l][r] = xc[b][l][:] . x_proj_w[r][:]  (K=256) ------
// One wave -> 1 M-tile x all 3 N-tiles.
__global__ void mamba_gemm_xproj(const _Float16* __restrict__ xc16,
                                 const _Float16* __restrict__ wxp,
                                 float* __restrict__ xdbl) {
  const int wave = blockIdx.x * 8 + (threadIdx.x >> 5);    // 16*64 waves
  const int lane = threadIdx.x & 31;
  const int b  = wave / 64;
  const int lt = wave % 64;
  const _Float16* A = xc16 + (size_t)b * NL * DI;
  v8f c[3] = {};
#pragma unroll
  for (int k = 0; k < DI; k += 32) {
    v16h a = load_a_frag(A, DI, lt * 16, k, lane);
#pragma unroll
    for (int j = 0; j < 3; ++j) {
      v16h bb = load_b_frag(wxp, DI, j * 16, k, lane);
      wmma_acc(c[j], a, bb);
    }
  }
  const int n = lane & 15, mb = (lane >> 4) * 8;
#pragma unroll
  for (int j = 0; j < 3; ++j) {
    float* dst = xdbl + ((size_t)b * NL + lt * 16 + mb) * RPAD + j * 16 + n;
#pragma unroll
    for (int i = 0; i < 8; ++i) dst[i * RPAD] = c[j][i];
  }
}

// ------------------------------ selective scan -----------------------------
// One block per batch; lane d in [0,256). State h[16] and A row in VGPRs.
// dt/B/C rows (48 floats, padded) staged through LDS via async global->LDS
// b128 copies, 32 steps per chunk.
__global__ void mamba_scan(const float* __restrict__ xdbl,
                           const float* __restrict__ xc32,
                           const _Float16* __restrict__ xz,
                           const float* __restrict__ dtw,
                           const float* __restrict__ dtb,
                           const float* __restrict__ Alog,
                           const float* __restrict__ Dp, int dir,
                           float* __restrict__ yacc,
                           _Float16* __restrict__ ysum) {
  const int b = blockIdx.x;
  const int d = threadIdx.x;                               // 256 threads
  float A[NST], h[NST], w[RNK];
#pragma unroll
  for (int n = 0; n < NST; ++n) { A[n] = -__expf(Alog[d * NST + n]); h[n] = 0.f; }
#pragma unroll
  for (int r = 0; r < RNK; ++r) w[r] = dtw[d * RNK + r];
  const float bias = dtb[d];
  const float Dd   = Dp[d];
  const _Float16* z = xz + ((size_t)b * E2 + DI + d) * NL; // z half of xz

  __shared__ __align__(16) float lds[32 * RPAD];
  for (int c = 0; c < NL / 32; ++c) {
    __syncthreads();   // previous chunk fully consumed before overwrite
#if HAVE_ASYNC_LDS
    // 32 rows x 12 b128 = 384 async copies, direct global->LDS (ASYNCcnt).
    for (int idx = threadIdx.x; idx < 32 * 12; idx += 256) {
      const int sj = idx / 12, q = idx % 12;
      const int s  = c * 32 + sj;
      const int li = dir ? (NL - 1 - s) : s;
      const float* src = &xdbl[((size_t)b * NL + li) * RPAD + q * 4];
      if (q == 0)  // warm L2/L0 for this step's u row (global_prefetch_b8)
        __builtin_prefetch(&xc32[((size_t)b * NL + li) * DI], 0, 1);
      __builtin_amdgcn_global_load_async_to_lds_b128(
          (__attribute__((address_space(1))) v4i*)const_cast<float*>(src),
          (__attribute__((address_space(3))) v4i*)&lds[sj * RPAD + q * 4],
          0, 0);
    }
#  if __has_builtin(__builtin_amdgcn_s_wait_asynccnt)
    __builtin_amdgcn_s_wait_asynccnt(0);
#  else
    asm volatile("s_wait_asynccnt 0" ::: "memory");
#  endif
#else
    for (int idx = threadIdx.x; idx < 32 * RPAD; idx += 256) {
      const int sj = idx / RPAD, f = idx % RPAD;
      const int s  = c * 32 + sj;
      const int li = dir ? (NL - 1 - s) : s;
      lds[idx] = xdbl[((size_t)b * NL + li) * RPAD + f];
    }
#endif
    __syncthreads();
#pragma unroll 4
    for (int sj = 0; sj < 32; ++sj) {
      const int s  = c * 32 + sj;
      const int li = dir ? (NL - 1 - s) : s;
      const float* row = &lds[sj * RPAD];
      float pre = bias;
#pragma unroll
      for (int r = 0; r < RNK; ++r) pre += row[r] * w[r];
      const float delta = (pre > 20.0f) ? pre : log1pf(__expf(pre));
      const float u  = xc32[((size_t)b * NL + li) * DI + d];
      const float du = delta * u;
      float y = 0.0f;
#pragma unroll
      for (int n = 0; n < NST; ++n) {
        h[n] = __expf(delta * A[n]) * h[n] + du * row[RNK + n];
        y += h[n] * row[RNK + NST + n];
      }
      y += Dd * u;
      const float zv = (float)z[li];
      y *= zv / (1.0f + __expf(-zv));                      // * silu(z)
      const size_t oi = ((size_t)b * NL + li) * DI + d;
      if (dir == 0) yacc[oi] = y;
      else          ysum[oi] = (_Float16)(yacc[oi] + y);
    }
  }
}

// ------ GEMM3: g3[b][l][t] = ysum[b][l][:] . out_proj_w[t][:]  (K=256) -----
// One wave -> 1 M-tile x 4 N-tiles.
__global__ void mamba_gemm_outproj(const _Float16* __restrict__ ysum,
                                   const _Float16* __restrict__ wout,
                                   float* __restrict__ g3) {
  const int wave = blockIdx.x * 8 + (threadIdx.x >> 5);    // 16*64*2 waves
  const int lane = threadIdx.x & 31;
  const int b  = wave / (64 * 2);
  const int rr = wave % (64 * 2);
  const int lt = rr / 2, ng = rr % 2;                      // ng: group of 4 t-tiles
  const _Float16* A = ysum + (size_t)b * NL * DI;
  v8f c[4] = {};
#pragma unroll
  for (int k = 0; k < DI; k += 32) {
    v16h a = load_a_frag(A, DI, lt * 16, k, lane);
#pragma unroll
    for (int j = 0; j < 4; ++j) {
      v16h bb = load_b_frag(wout, DI, (ng * 4 + j) * 16, k, lane);
      wmma_acc(c[j], a, bb);
    }
  }
  const int n = lane & 15, mb = (lane >> 4) * 8;
#pragma unroll
  for (int j = 0; j < 4; ++j) {
    float* dst =
        g3 + ((size_t)b * NL + lt * 16 + mb) * NT + (ng * 4 + j) * 16 + n;
#pragma unroll
    for (int i = 0; i < 8; ++i) dst[i * NT] = c[j][i];
  }
}

// -------------- residual + final rmsnorm + transpose back ------------------
__global__ void mamba_rmsnorm_out(const float* __restrict__ g3,
                                  const float* __restrict__ hid,
                                  const float* __restrict__ nfw,
                                  float* __restrict__ out) {
  const int l = blockIdx.x & (NL - 1);
  const int b = blockIdx.x >> 10;
  const int t = threadIdx.x;                               // 128 threads
  float g = g3[((size_t)b * NL + l) * NT + t] +
            hid[((size_t)b * NT + t) * NL + l];
  float ss = g * g;
#pragma unroll
  for (int o = 16; o > 0; o >>= 1) ss += __shfl_xor(ss, o, 32);
  __shared__ float sred[4];
  if ((t & 31) == 0) sred[t >> 5] = ss;
  __syncthreads();
  float tot = sred[0] + sred[1] + sred[2] + sred[3];
  float inv = rsqrtf(tot * (1.0f / NT) + 1e-5f);
  out[((size_t)b * NT + t) * NL + l] = g * inv * nfw[t];
}

// ---------------------------------------------------------------------------
extern "C" void kernel_launch(void* const* d_in, const int* in_sizes, int n_in,
                              void* d_out, int out_size, void* d_ws, size_t ws_size,
                              hipStream_t stream) {
  const float* hid    = (const float*)d_in[0];   // (16,128,1024)
  const float* norm_w = (const float*)d_in[1];   // (128,)
  const float* winp   = (const float*)d_in[2];   // (512,128)
  const float* conv_w = (const float*)d_in[3];   // (256,4)
  const float* conv_b = (const float*)d_in[4];
  const float* xproj  = (const float*)d_in[5];   // (40,256)
  const float* dtw    = (const float*)d_in[6];   // (256,8)
  const float* dtb    = (const float*)d_in[7];
  const float* Alog   = (const float*)d_in[8];   // (256,16)
  const float* Dp     = (const float*)d_in[9];
  const float* conv_wb= (const float*)d_in[10];
  const float* conv_bb= (const float*)d_in[11];
  const float* xprojb = (const float*)d_in[12];
  const float* dtwb   = (const float*)d_in[13];
  const float* dtbb   = (const float*)d_in[14];
  const float* Alogb  = (const float*)d_in[15];
  const float* Dpb    = (const float*)d_in[16];
  const float* woutp  = (const float*)d_in[17];  // (128,256)
  const float* nfw    = (const float*)d_in[18];
  float* out = (float*)d_out;

  char* ws = (char*)d_ws;
  _Float16* win16  = (_Float16*)(ws + OFF_WIN16);
  _Float16* xpf16  = (_Float16*)(ws + OFF_WXPF16);
  _Float16* xpb16  = (_Float16*)(ws + OFF_WXPB16);
  _Float16* wout16 = (_Float16*)(ws + OFF_WOUT16);
  _Float16* hn16   = (_Float16*)(ws + OFF_HN16);
  _Float16* xz16   = (_Float16*)(ws + OFF_XZ16);
  _Float16* xc16   = (_Float16*)(ws + OFF_XC16);
  float*    xc32   = (float*)   (ws + OFF_XC32);
  float*    xdbl   = (float*)   (ws + OFF_XDBL);
  float*    yacc   = (float*)   (ws + OFF_YACC);
  _Float16* ysum16 = (_Float16*)(ws + OFF_YSUM16);
  float*    g3     = (float*)   (ws + OFF_G3);

  // 1) f16 weight copies (x_proj padded 40->48 rows with zeros)
  mamba_cvt_weights<<<480, 256, 0, stream>>>(winp, xproj, xprojb, woutp,
                                             win16, xpf16, xpb16, wout16);
  // 2) rmsnorm + transpose -> hn f16 [b][l][t]
  mamba_rmsnorm_in<<<NB * NL, NT, 0, stream>>>(hid, norm_w, hn16);
  // 3) in_proj GEMM -> xz f16 [b][e][l]
  mamba_gemm_inproj<<<1024, 256, 0, stream>>>(hn16, win16, xz16);

  // ---- forward direction ----
  mamba_conv_silu<<<NB * DI, 256, 0, stream>>>(xz16, conv_w, conv_b, 0, xc16, xc32);
  mamba_gemm_xproj<<<128, 256, 0, stream>>>(xc16, xpf16, xdbl);
  mamba_scan<<<NB, DI, 0, stream>>>(xdbl, xc32, xz16, dtw, dtb, Alog, Dp, 0,
                                    yacc, ysum16);
  // ---- backward direction ----
  mamba_conv_silu<<<NB * DI, 256, 0, stream>>>(xz16, conv_wb, conv_bb, 1, xc16, xc32);
  mamba_gemm_xproj<<<128, 256, 0, stream>>>(xc16, xpb16, xdbl);
  mamba_scan<<<NB, DI, 0, stream>>>(xdbl, xc32, xz16, dtwb, dtbb, Alogb, Dpb, 1,
                                    yacc, ysum16);

  // 4) out_proj GEMM -> g3 f32 [b][l][t]
  mamba_gemm_outproj<<<256, 256, 0, stream>>>(ysum16, wout16, g3);
  // 5) residual + final rmsnorm + transpose back -> (B,T,DM)
  mamba_rmsnorm_out<<<NB * NL, NT, 0, stream>>>(g3, hid, nfw, out);
}
